// TransformerBlock_19705309954286
// MI455X (gfx1250) — matrix operand order repaired and checked by reading, not compiled, not run on hardware
//
#include <hip/hip_runtime.h>
#include <cstdint>
#include <cstddef>

typedef __bf16 bf16;
typedef __attribute__((ext_vector_type(8)))  __bf16 v8bf;
typedef __attribute__((ext_vector_type(16))) __bf16 v16bf;
typedef __attribute__((ext_vector_type(8)))  float  v8f;
typedef __attribute__((ext_vector_type(4)))  unsigned int u32x4;
typedef __attribute__((ext_vector_type(8)))  int i32x8;
typedef __attribute__((ext_vector_type(4)))  int i32x4;

#define TB_B  2
#define TB_S  2048
#define TB_D  1024
#define TB_H  16
#define TB_HD 64
#define TB_F  4096
#define TB_M  (TB_B * TB_S)

#define GEMM_BK   64
#define LDS_STR   72   // 64-elem row + 8-elem TDM pad -> 144B stride, conflict-free

static __device__ inline v16bf cat16(v8bf lo, v8bf hi) {
  return __builtin_shufflevector(lo, hi, 0,1,2,3,4,5,6,7,8,9,10,11,12,13,14,15);
}
static __device__ inline v8f wmma_bf16(v16bf a, v16bf b, v8f c) {
  // D = A(16x32 bf16) * B(32x16 bf16) + C(16x16 f32)
  return __builtin_amdgcn_wmma_f32_16x16x32_bf16(false, a, false, b, (short)0, c,
                                                 false, false);
}

// ---------------------------------------------------------------------------
// TDM: 2D tensor tile -> LDS. data_size = 2 bytes (bf16).
// Tile tile1 rows x tile0 cols out of a (tdim1 x tdim0) row-major tensor with
// row stride stride0 (elements). LDS rows padded: 32 DWORDs data + 4 DWORDs
// pad => LDS row stride = 72 bf16 elements (requires tile0 == 64).
// ---------------------------------------------------------------------------
static __device__ inline void tdm_load_2d(const void* gptr, unsigned lds_addr,
                                          unsigned tile0, unsigned tile1,
                                          unsigned long long tdim0,
                                          unsigned long long tdim1,
                                          unsigned long long stride0) {
  unsigned long long ga = (unsigned long long)(uintptr_t)gptr;
  u32x4 g0;
  g0.x = 1u;                                      // count=1 (valid user D#)
  g0.y = lds_addr;                                // lds_addr [63:32]
  g0.z = (unsigned)(ga & 0xffffffffu);            // global_addr lo
  g0.w = (unsigned)((ga >> 32) & 0x1ffffffu)      // global_addr [56:32]
         | (2u << 30);                            // type=2 ("image")
  // group1: wg_mask=0 | data_size=1(2B)<<16 | pad_enable<<20 |
  //         pad_interval=4 (32 DW) <<22 | pad_amount=3 (4 DW) <<25
  unsigned d0 = (1u << 16) | (1u << 20) | (4u << 22) | (3u << 25);
  unsigned d1 = ((unsigned)(tdim0 & 0xffffu)) << 16;            // [63:48]=td0.lo16
  unsigned d2 = (unsigned)((tdim0 >> 16) & 0xffffu)             // [79:64]=td0.hi16
              | ((unsigned)(tdim1 & 0xffffu) << 16);            // [95:80]=td1.lo16
  unsigned d3 = (unsigned)((tdim1 >> 16) & 0xffffu)             // [111:96]=td1.hi16
              | (tile0 << 16);                                  // [127:112]=tile0
  unsigned d4 = (tile1 & 0xffffu);                              // tile1, tile2=0
  unsigned d5 = (unsigned)(stride0 & 0xffffffffu);              // dim0 stride lo
  unsigned d6 = (unsigned)((stride0 >> 32) & 0xffffu);          // stride hi, s1=0
  unsigned d7 = 0;
  i32x8 g1 = { (int)d0, (int)d1, (int)d2, (int)d3,
               (int)d4, (int)d5, (int)d6, (int)d7 };
  i32x4 z4 = { 0, 0, 0, 0 };
  i32x8 z8 = { 0, 0, 0, 0, 0, 0, 0, 0 };
  __builtin_amdgcn_tensor_load_to_lds(g0, g1, z4, z4, z8, 0);
}

// ---------------------------------------------------------------------------
// f32 [K,N] -> bf16 transposed [N,K] (weights; makes GEMM B-tiles row-major)
// 64x64 tiles, both phases coalesced via LDS staging
// ---------------------------------------------------------------------------
__global__ __launch_bounds__(256)
void cvt_transpose_bf16(const float* __restrict__ in, bf16* __restrict__ outT,
                        int K, int N) {
  __shared__ bf16 t[64][65];
  const int tid = threadIdx.x;
  const int n0 = blockIdx.x * 64;
  const int k0 = blockIdx.y * 64;
#pragma unroll
  for (int it = 0; it < 16; ++it) {
    int i = tid + it * 256;
    int kk = i >> 6, nn = i & 63;  // read coalesced along n
    t[nn][kk] = (bf16)in[(size_t)(k0 + kk) * N + n0 + nn];
  }
  __syncthreads();
#pragma unroll
  for (int it = 0; it < 16; ++it) {
    int i = tid + it * 256;
    int nn = i >> 6, kk = i & 63;  // write coalesced along k
    outT[(size_t)(n0 + nn) * K + k0 + kk] = t[nn][kk];
  }
}

// ---------------------------------------------------------------------------
// LayerNorm (torch-style unbiased variance, ddof=1) -> bf16 output
// ---------------------------------------------------------------------------
__global__ __launch_bounds__(256)
void layernorm_to_bf16(const float* __restrict__ x, const float* __restrict__ g,
                       const float* __restrict__ bb, bf16* __restrict__ y) {
  __shared__ float sh[256];
  const int row = blockIdx.x;
  const int tid = threadIdx.x;
  const float* xr = x + (size_t)row * TB_D;
  float s = 0.f, s2 = 0.f;
  for (int i = tid; i < TB_D; i += 256) { float v = xr[i]; s += v; s2 += v * v; }
  sh[tid] = s; __syncthreads();
  for (int off = 128; off > 0; off >>= 1) {
    if (tid < off) sh[tid] += sh[tid + off];
    __syncthreads();
  }
  float sum = sh[0]; __syncthreads();
  sh[tid] = s2; __syncthreads();
  for (int off = 128; off > 0; off >>= 1) {
    if (tid < off) sh[tid] += sh[tid + off];
    __syncthreads();
  }
  float sum2 = sh[0];
  float mean = sum * (1.0f / TB_D);
  float var  = (sum2 - sum * mean) * (1.0f / (TB_D - 1));  // unbiased
  float rstd = rsqrtf(var + 1e-5f);
  bf16* yr = y + (size_t)row * TB_D;
  for (int i = tid; i < TB_D; i += 256)
    yr[i] = (bf16)(g[i] * ((xr[i] - mean) * rstd) + bb[i]);
}

// ---------------------------------------------------------------------------
// Tiled bf16 WMMA GEMM with TDM double-buffered staging:
//   C[M,N] = A[M,K] * Wt[N,K]^T (+bias)(+gelu)(+resid)
// Wt is the transposed weight [N,K] row-major, so both A and B tiles are
// plain 2D row tiles for the Tensor Data Mover, and both fragment types are
// contiguous 16B LDS reads. Block tile 128x64, 8 waves, wave tile 32x32.
// ---------------------------------------------------------------------------
__global__ __launch_bounds__(256)
void gemm_bf16_wmma(const bf16* __restrict__ A, const bf16* __restrict__ Wt,
                    const float* __restrict__ bias, const float* __restrict__ resid,
                    float* __restrict__ outF, bf16* __restrict__ outB,
                    int M, int N, int K, int doGelu) {
  __shared__ __align__(16) bf16 sA[2][128 * LDS_STR];  // 128 M rows x 64 K
  __shared__ __align__(16) bf16 sB[2][64 * LDS_STR];   // 64 N rows x 64 K
  const int tid  = threadIdx.x;
  const int lane = tid & 31;
  const int wave = tid >> 5;
  const int wm = (wave & 3) * 32;
  const int wn = (wave >> 2) * 32;
  const int lm = lane & 15;
  const int lh = lane >> 4;
  const int blockN = blockIdx.x * 64;
  const int blockM = blockIdx.y * 128;
  const int nk = K / GEMM_BK;

  v8f acc[2][2] = {};

  if (wave == 0) {
    tdm_load_2d(A + (size_t)blockM * K, (unsigned)(uintptr_t)&sA[0][0],
                GEMM_BK, 128, (unsigned)K, (unsigned)M, (unsigned)K);
    tdm_load_2d(Wt + (size_t)blockN * K, (unsigned)(uintptr_t)&sB[0][0],
                GEMM_BK, 64, (unsigned)K, (unsigned)N, (unsigned)K);
  }

  for (int kt = 0; kt < nk; ++kt) {
    const int cur = kt & 1;
    if (wave == 0) {
      if (kt + 1 < nk) {  // prefetch next stage into the other buffer
        const int k1 = (kt + 1) * GEMM_BK;
        tdm_load_2d(A + (size_t)blockM * K + k1,
                    (unsigned)(uintptr_t)&sA[cur ^ 1][0],
                    GEMM_BK, 128, (unsigned)K, (unsigned)M, (unsigned)K);
        tdm_load_2d(Wt + (size_t)blockN * K + k1,
                    (unsigned)(uintptr_t)&sB[cur ^ 1][0],
                    GEMM_BK, 64, (unsigned)K, (unsigned)N, (unsigned)K);
        __builtin_amdgcn_s_wait_tensorcnt(2);  // current stage's 2 ops done
      } else {
        __builtin_amdgcn_s_wait_tensorcnt(0);
      }
    }
    __syncthreads();

    const bf16* a   = sA[cur];
    const bf16* bsh = sB[cur];
#pragma unroll
    for (int kc = 0; kc < 2; ++kc) {  // two 32-wide K chunks
      v16bf af[2], bfg[2];
#pragma unroll
      for (int t = 0; t < 2; ++t) {
        // A frag: lane row m=lm; elems i: K=kc*32+lh*16+i (contiguous)
        const v8bf* ap = (const v8bf*)(a + (wm + t * 16 + lm) * LDS_STR + kc * 32);
        af[t] = cat16(ap[2 * lh], ap[2 * lh + 1]);
        // B frag: lane col n=lm; elems i: K=kc*32+lh*16+i (contiguous in Wt row)
        const v8bf* bp = (const v8bf*)(bsh + (wn + t * 16 + lm) * LDS_STR +
                                       kc * 32 + lh * 16);
        bfg[t] = cat16(bp[0], bp[1]);
      }
#pragma unroll
      for (int am = 0; am < 2; ++am)
#pragma unroll
        for (int bn = 0; bn < 2; ++bn)
          acc[am][bn] = wmma_bf16(af[am], bfg[bn], acc[am][bn]);
    }
    __syncthreads();
  }

  // epilogue: C layout lane->(n=lm), vgpr r->(m = r + 8*lh)
#pragma unroll
  for (int am = 0; am < 2; ++am)
#pragma unroll
    for (int bn = 0; bn < 2; ++bn)
#pragma unroll
      for (int r = 0; r < 8; ++r) {
        int gm = blockM + wm + am * 16 + r + 8 * lh;
        int gn = blockN + wn + bn * 16 + lm;
        size_t o = (size_t)gm * N + gn;
        float v = acc[am][bn][r];
        if (bias) v += bias[gn];
        if (doGelu) {
          float u = 0.7978845608028654f * (v + 0.044715f * v * v * v);
          v = 0.5f * v * (1.0f + tanhf(u));
        }
        if (resid) v += resid[o];
        if (outF) outF[o] = v;
        if (outB) outB[o] = (bf16)v;
      }
}

// ---------------------------------------------------------------------------
// Flash attention (causal), one wave per 16-query tile per (b,h)
// ---------------------------------------------------------------------------
__global__ __launch_bounds__(128)
void flash_attn_wmma(const bf16* __restrict__ qkv, bf16* __restrict__ out) {
  __shared__ __align__(16) bf16 sP[4 * 16 * 40];  // per-wave P staging
  const int tid  = threadIdx.x;
  const int lane = tid & 31;
  const int wv   = tid >> 5;
  const int lm = lane & 15;
  const int lh = lane >> 4;
  const int bh = blockIdx.y;
  const int b = bh / TB_H;
  const int h = bh % TB_H;
  const int q0 = (blockIdx.x * 4 + wv) * 16;
  const size_t rs = 3 * TB_D;
  const bf16* qb = qkv + (size_t)(b * TB_S + q0) * rs + h * TB_HD;
  const bf16* kb = qkv + (size_t)(b * TB_S) * rs + TB_D + h * TB_HD;
  const bf16* vb = qkv + (size_t)(b * TB_S) * rs + 2 * TB_D + h * TB_HD;
  bf16* P = sP + wv * (16 * 40);

  v16bf aq[2];
#pragma unroll
  for (int t = 0; t < 2; ++t) {
    const v8bf* p0 = (const v8bf*)(qb + (size_t)lm * rs + t * 32 + lh * 16);
    const v8bf* p1 = (const v8bf*)(qb + (size_t)lm * rs + t * 32 + lh * 16 + 8);
    aq[t] = cat16(*p0, *p1);
  }

  v8f o[4] = {};
  float mrow[8], lrow[8];
#pragma unroll
  for (int r = 0; r < 8; ++r) { mrow[r] = -1e30f; lrow[r] = 0.f; }

  for (int k0 = 0; k0 < q0 + 16; k0 += 32) {
    v8f s[2];
#pragma unroll
    for (int nt = 0; nt < 2; ++nt) {
      const bf16* kr = kb + (size_t)(k0 + nt * 16 + lm) * rs;
      const v8bf* kp0 = (const v8bf*)(kr + 0 * 32 + lh * 16);
      const v8bf* kp1 = (const v8bf*)(kr + 1 * 32 + lh * 16);
      v16bf bk0 = cat16(kp0[0], kp0[1]);
      v16bf bk1 = cat16(kp1[0], kp1[1]);
      v8f z = {};
      s[nt] = wmma_bf16(aq[0], bk0, z);
      s[nt] = wmma_bf16(aq[1], bk1, s[nt]);
    }
    float bm[8];
#pragma unroll
    for (int r = 0; r < 8; ++r) {
      int qm = q0 + r + 8 * lh;
#pragma unroll
      for (int nt = 0; nt < 2; ++nt) {
        int kn = k0 + nt * 16 + lm;
        float sv = s[nt][r] * 0.125f;  // 1/sqrt(64)
        s[nt][r] = (kn <= qm) ? sv : -1e30f;
      }
      bm[r] = fmaxf(s[0][r], s[1][r]);
    }
#pragma unroll
    for (int r = 0; r < 8; ++r)
#pragma unroll
      for (int mk = 1; mk < 16; mk <<= 1)
        bm[r] = fmaxf(bm[r], __shfl_xor(bm[r], mk, 32));
    float scl[8];
#pragma unroll
    for (int r = 0; r < 8; ++r) {
      float mnew = fmaxf(mrow[r], bm[r]);
      scl[r] = __expf(mrow[r] - mnew);
      mrow[r] = mnew;
    }
    float ps[8];
#pragma unroll
    for (int r = 0; r < 8; ++r) {
      s[0][r] = __expf(s[0][r] - mrow[r]);
      s[1][r] = __expf(s[1][r] - mrow[r]);
      ps[r] = s[0][r] + s[1][r];
    }
#pragma unroll
    for (int r = 0; r < 8; ++r)
#pragma unroll
      for (int mk = 1; mk < 16; mk <<= 1)
        ps[r] += __shfl_xor(ps[r], mk, 32);
#pragma unroll
    for (int r = 0; r < 8; ++r) lrow[r] = lrow[r] * scl[r] + ps[r];
#pragma unroll
    for (int g = 0; g < 4; ++g)
#pragma unroll
      for (int r = 0; r < 8; ++r) o[g][r] *= scl[r];

#pragma unroll
    for (int nt = 0; nt < 2; ++nt)
#pragma unroll
      for (int r = 0; r < 8; ++r)
        P[(r + 8 * lh) * 40 + nt * 16 + lm] = (bf16)s[nt][r];
    asm volatile("s_wait_dscnt 0" ::: "memory");  // same-wave LDS RAW
    const v8bf* pp = (const v8bf*)&P[lm * 40];
    v16bf ap = cat16(pp[2 * lh], pp[2 * lh + 1]);

#pragma unroll
    for (int g = 0; g < 4; ++g) {
      v16bf bv;
#pragma unroll
      for (int i = 0; i < 16; ++i)
        bv[i] = vb[(size_t)(k0 + lh * 16 + i) * rs + g * 16 + lm];
      o[g] = wmma_bf16(ap, bv, o[g]);
    }
  }

#pragma unroll
  for (int g = 0; g < 4; ++g)
#pragma unroll
    for (int r = 0; r < 8; ++r) {
      int qm = q0 + r + 8 * lh;
      out[(size_t)(b * TB_S + qm) * TB_D + h * TB_HD + g * 16 + lm] =
          (bf16)(o[g][r] / lrow[r]);
    }
}

// ---------------------------------------------------------------------------
// launcher
// ---------------------------------------------------------------------------
extern "C" void kernel_launch(void* const* d_in, const int* in_sizes, int n_in,
                              void* d_out, int out_size, void* d_ws, size_t ws_size,
                              hipStream_t stream) {
  (void)in_sizes; (void)n_in; (void)out_size; (void)ws_size;
  const float* x      = (const float*)d_in[0];
  const float* ln1_g  = (const float*)d_in[1];
  const float* ln1_b  = (const float*)d_in[2];
  const float* w_attn = (const float*)d_in[3];
  const float* b_attn = (const float*)d_in[4];
  const float* w_ao   = (const float*)d_in[5];
  const float* b_ao   = (const float*)d_in[6];
  const float* ln2_g  = (const float*)d_in[7];
  const float* ln2_b  = (const float*)d_in[8];
  const float* w_fc   = (const float*)d_in[9];
  const float* b_fc   = (const float*)d_in[10];
  const float* w_pr   = (const float*)d_in[11];
  const float* b_pr   = (const float*)d_in[12];
  float* out = (float*)d_out;

  char* ws = (char*)d_ws;
  size_t off = 0;
  auto alloc = [&](size_t bytes) {
    void* p = ws + off;
    off += (bytes + 255) & ~(size_t)255;
    return p;
  };
  bf16*  w_attn_t = (bf16*)alloc((size_t)TB_D * 3 * TB_D * 2);  // [3D, D]
  bf16*  w_ao_t   = (bf16*)alloc((size_t)TB_D * TB_D * 2);      // [D, D]
  bf16*  w_fc_t   = (bf16*)alloc((size_t)TB_D * TB_F * 2);      // [F, D]
  bf16*  w_pr_t   = (bf16*)alloc((size_t)TB_F * TB_D * 2);      // [D, F]
  bf16*  ln1o     = (bf16*)alloc((size_t)TB_M * TB_D * 2);
  bf16*  qkv      = (bf16*)alloc((size_t)TB_M * 3 * TB_D * 2);
  bf16*  attn     = (bf16*)alloc((size_t)TB_M * TB_D * 2);
  float* x1       = (float*)alloc((size_t)TB_M * TB_D * 4);
  bf16*  ln2o     = (bf16*)alloc((size_t)TB_M * TB_D * 2);
  bf16*  hbuf     = (bf16*)alloc((size_t)TB_M * TB_F * 2);

  auto cvtT = [&](const float* src, bf16* dstT, int K, int N) {
    cvt_transpose_bf16<<<dim3(N / 64, K / 64), 256, 0, stream>>>(src, dstT, K, N);
  };
  cvtT(w_attn, w_attn_t, TB_D, 3 * TB_D);
  cvtT(w_ao,   w_ao_t,   TB_D, TB_D);
  cvtT(w_fc,   w_fc_t,   TB_D, TB_F);
  cvtT(w_pr,   w_pr_t,   TB_F, TB_D);

  layernorm_to_bf16<<<TB_M, 256, 0, stream>>>(x, ln1_g, ln1_b, ln1o);

  gemm_bf16_wmma<<<dim3(3 * TB_D / 64, TB_M / 128), 256, 0, stream>>>(
      ln1o, w_attn_t, b_attn, nullptr, nullptr, qkv, TB_M, 3 * TB_D, TB_D, 0);

  flash_attn_wmma<<<dim3(TB_S / 64, TB_B * TB_H), 128, 0, stream>>>(qkv, attn);

  gemm_bf16_wmma<<<dim3(TB_D / 64, TB_M / 128), 256, 0, stream>>>(
      attn, w_ao_t, b_ao, x, x1, nullptr, TB_M, TB_D, TB_D, 0);

  layernorm_to_bf16<<<TB_M, 256, 0, stream>>>(x1, ln2_g, ln2_b, ln2o);

  gemm_bf16_wmma<<<dim3(TB_F / 64, TB_M / 128), 256, 0, stream>>>(
      ln2o, w_fc_t, b_fc, nullptr, nullptr, hbuf, TB_M, TB_F, TB_D, 1);

  gemm_bf16_wmma<<<dim3(TB_D / 64, TB_M / 128), 256, 0, stream>>>(
      hbuf, w_pr_t, b_pr, x1, out, nullptr, TB_M, TB_D, TB_F, 0);
}
